// SpectralBottleneckLM_49589692400200
// MI455X (gfx1250) — compile-verified, matrix operand order checked by reading, not run on hardware
//
#include <hip/hip_runtime.h>
#include <math.h>
#include <stdint.h>

// ---------------------------------------------------------------------------
// SpectralBottleneckLM forward for gfx1250 (MI455X), WMMA f16->f32 GEMM core.
// B=2, T=2048 (S=4096 rows), D=384, NF=128, BD=192, NL=6, V=50257.
//
// GEMM v3:
//  - NT path: 4 waves/block, 64x64 block tile. B K-chunk (32x64 f32) is moved
//    global->LDS by the Tensor Data Mover (tensor_load_to_lds, TENSORcnt),
//    double-buffered so DMA of chunk k+1 overlaps WMMA on chunk k.
//  - Each wave reuses its A fragment across a 16x64 strip (4 accumulators).
//  - TB path (logits vs emb^T): K-invariant bounds guard hoisted out of the
//    K loop (full-tile fast path; 785/786 strips take it).
// ---------------------------------------------------------------------------

typedef __attribute__((ext_vector_type(16))) _Float16 v16h;
typedef __attribute__((ext_vector_type(8)))  float    v8f;
typedef __attribute__((ext_vector_type(4)))  unsigned u32x4;
typedef __attribute__((ext_vector_type(8)))  int      i32x8;
typedef __attribute__((ext_vector_type(4)))  int      i32x4;

#define DIM_D  384
#define DIM_NF 128
#define DIM_BD 192
#define DIM_V  50257
#define SEQ_B  2
#define SEQ_T  2048
#define ROWS   (SEQ_B * SEQ_T)       // 4096
#define N_LAYERS 6

#if defined(__has_builtin)
#  if __has_builtin(__builtin_amdgcn_tensor_load_to_lds) && \
      __has_builtin(__builtin_amdgcn_s_wait_tensorcnt)
#    define HAVE_TDM 1
#  endif
#endif
#ifndef HAVE_TDM
#  define HAVE_TDM 0
#endif

__device__ __forceinline__ float gelu_exact(float x) {
    return 0.5f * x * (1.0f + erff(x * 0.70710678118654752f));
}
__device__ __forceinline__ float sigmoidf(float x) {
    return 1.0f / (1.0f + expf(-x));
}

#if HAVE_TDM
// Issue a 2D TDM load: tile (tile_d1 rows x tile_d0 elems of 4B) from a
// tensor with row stride stride_d0 (elems) at gaddr, into LDS at lds_off.
// D# packing per CDNA5 ISA 8.3/8.4 (count=1, type=2="image", data_size=2->4B).
// This toolchain's builtin takes 6 args (g0,g1,g2,g3,g4,cpol).
__device__ __forceinline__ void tdm_load_2d(unsigned lds_off, const float* gaddr,
                                            unsigned tensor_d0, unsigned tensor_d1,
                                            unsigned tile_d0, unsigned tile_d1,
                                            unsigned stride_d0)
{
    const unsigned long long ga = (unsigned long long)(uintptr_t)gaddr;
    u32x4 g0;
    g0[0] = 1u;                                          // count=1, user mode
    g0[1] = lds_off;                                     // lds_addr (bytes)
    g0[2] = (unsigned)(ga & 0xFFFFFFFFu);                // global_addr[31:0]
    g0[3] = (unsigned)((ga >> 32) & 0x01FFFFFFu)         // global_addr[56:32]
          | (2u << 30);                                  // type=2
    i32x8 g1;
    g1[0] = (int)(2u << 16);                             // data_size=2 (4B)
    g1[1] = (int)((tensor_d0 & 0xFFFFu) << 16);          // tensor_dim0[15:0]
    g1[2] = (int)(((tensor_d0 >> 16) & 0xFFFFu) | ((tensor_d1 & 0xFFFFu) << 16));
    g1[3] = (int)(((tensor_d1 >> 16) & 0xFFFFu) | ((tile_d0 & 0xFFFFu) << 16));
    g1[4] = (int)(tile_d1 & 0xFFFFu);                    // tile_dim1, tile_dim2=0
    g1[5] = (int)stride_d0;                              // tensor_dim0_stride lo
    g1[6] = 0;
    g1[7] = 0;
    const i32x4 z4 = {0, 0, 0, 0};                       // groups 2/3 unused (2D)
    const i32x8 z8 = {0, 0, 0, 0, 0, 0, 0, 0};
    __builtin_amdgcn_tensor_load_to_lds(g0, g1, z4, z4, z8, 0);
}
#endif

// A-matrix f16 lane layout (ISA 7.12.2): lane L holds row M=L%16; hf=L/16:
//   element e -> k = k0 + (e<8?0:16) + (e&7) + hf*8.
// B fragment: lane L holds column N=L%16 with the same K element pattern.
// C/D (v8f): lane L: n = L%16, vgpr r -> m = r + 8*(L/16).

__device__ __forceinline__ v16h load_a_frag(const float* arow, int k0, int hf) {
    float af[16];
    const float* ap = arow + k0 + hf * 8;
    *(float4*)(af + 0)  = *(const float4*)(ap + 0);
    *(float4*)(af + 4)  = *(const float4*)(ap + 4);
    *(float4*)(af + 8)  = *(const float4*)(ap + 16);
    *(float4*)(af + 12) = *(const float4*)(ap + 20);
    v16h av;
#pragma unroll
    for (int e = 0; e < 16; ++e) av[e] = (_Float16)af[e];
    return av;
}

// ---------------------------------------------------------------------------
// Non-transposed WMMA GEMM: C[M,N] = act(A[M,K] @ W[K,N] + bias) (+ resid)
// Requires M%64==0, N%64==0, K%32==0 (true for all intra-network GEMMs).
// ---------------------------------------------------------------------------
template<int ACT, bool RESID>
__global__ void wmma_gemm_nt_kernel(const float* __restrict__ A,
                                    const float* __restrict__ W,
                                    const float* __restrict__ bias,
                                    const float* __restrict__ resid,
                                    float* __restrict__ C,
                                    int M, int N, int K)
{
    __shared__ float bshf[2][32 * 64];          // double-buffered B chunk (f32)

    const int tid  = threadIdx.x;               // 0..127
    const int lane = tid & 31;
    const int wave = tid >> 5;                  // 0..3
    const int hf   = lane >> 4;                 // 0/1
    const int l16  = lane & 15;

    const int blocksN = N >> 6;
    const int bM = blockIdx.x / blocksN;
    const int bN = blockIdx.x - bM * blocksN;
    const int mBase = bM * 64;
    const int nBase = bN * 64;

    const int rowA = mBase + wave * 16 + l16;
    const float* arow = A + (size_t)rowA * K;

    v8f acc[4] = {v8f{}, v8f{}, v8f{}, v8f{}};
    const int nk = K >> 5;

#if HAVE_TDM
    // ---- async pipeline: TDM moves B chunks, wave 0 manages TENSORcnt
    if (wave == 0)
        tdm_load_2d((unsigned)(uintptr_t)&bshf[0][0], W + (size_t)0 * N + nBase,
                    (unsigned)N, (unsigned)K, 64u, 32u, (unsigned)N);
    int cur = 0;
    for (int kc = 0; kc < nk; ++kc) {
        if (wave == 0) __builtin_amdgcn_s_wait_tensorcnt(0);
        __syncthreads();                        // buf[cur] ready; buf[cur^1] free
        if (wave == 0 && kc + 1 < nk)
            tdm_load_2d((unsigned)(uintptr_t)&bshf[cur ^ 1][0],
                        W + (size_t)(kc + 1) * 32 * N + nBase,
                        (unsigned)N, (unsigned)K, 64u, 32u, (unsigned)N);

        const v16h av = load_a_frag(arow, kc * 32, hf);
        const float* bb = &bshf[cur][0];
#pragma unroll
        for (int nt = 0; nt < 4; ++nt) {
            const int nLoc = nt * 16 + l16;
            v16h bv;
#pragma unroll
            for (int e = 0; e < 16; ++e) {
                const int k = ((e & 8) ? 16 : 0) + (e & 7) + hf * 8;
                bv[e] = (_Float16)bb[k * 64 + nLoc];
            }
            acc[nt] = __builtin_amdgcn_wmma_f32_16x16x32_f16(
                          false, av, false, bv, (short)0, acc[nt], false, false);
        }
        cur ^= 1;
    }
#else
    // ---- fallback: synchronous coalesced staging
    for (int kc = 0; kc < nk; ++kc) {
        const int k0 = kc * 32;
        __syncthreads();
#pragma unroll
        for (int i = 0; i < 4; ++i) {
            const int linear = tid + i * 128;
            const int r  = linear >> 4;
            const int c4 = linear & 15;
            *(float4*)(&bshf[0][r * 64 + c4 * 4]) =
                *(const float4*)(W + (size_t)(k0 + r) * N + nBase + c4 * 4);
        }
        __syncthreads();
        const v16h av = load_a_frag(arow, k0, hf);
#pragma unroll
        for (int nt = 0; nt < 4; ++nt) {
            const int nLoc = nt * 16 + l16;
            v16h bv;
#pragma unroll
            for (int e = 0; e < 16; ++e) {
                const int k = ((e & 8) ? 16 : 0) + (e & 7) + hf * 8;
                bv[e] = (_Float16)bshf[0][k * 64 + nLoc];
            }
            acc[nt] = __builtin_amdgcn_wmma_f32_16x16x32_f16(
                          false, av, false, bv, (short)0, acc[nt], false, false);
        }
    }
#endif

    // ---- epilogue
#pragma unroll
    for (int nt = 0; nt < 4; ++nt) {
        const int n = nBase + nt * 16 + l16;
#pragma unroll
        for (int r = 0; r < 8; ++r) {
            const int m = mBase + wave * 16 + r + hf * 8;
            float v = acc[nt][r];
            if (bias) v += bias[n];
            if (ACT == 1) v = gelu_exact(v);
            else if (ACT == 2) v = sigmoidf(v);
            if (RESID) v += resid[(size_t)m * N + n];
            C[(size_t)m * N + n] = v;
        }
    }
}

// ---------------------------------------------------------------------------
// Transposed-B WMMA GEMM for logits: C[M,V] = A[M,K] @ emb[V,K]^T
// Each wave owns a 16x64 strip. Bounds guard hoisted out of the K loop.
// ---------------------------------------------------------------------------
template<bool FULL>
__device__ __forceinline__ void tb_accumulate(const float* arow,
                                              const float* __restrict__ W,
                                              int nStrip, int N, int K,
                                              int hf, int l16, v8f acc[4])
{
    for (int k0 = 0; k0 < K; k0 += 32) {
        const v16h av = load_a_frag(arow, k0, hf);
#pragma unroll
        for (int nt = 0; nt < 4; ++nt) {
            const int colB = nStrip + nt * 16 + l16;
            v16h bv;
            if (FULL || colB < N) {
                float bf[16];
                const float* bp = W + (size_t)colB * K + k0 + hf * 8;
                *(float4*)(bf + 0)  = *(const float4*)(bp + 0);
                *(float4*)(bf + 4)  = *(const float4*)(bp + 4);
                *(float4*)(bf + 8)  = *(const float4*)(bp + 16);
                *(float4*)(bf + 12) = *(const float4*)(bp + 20);
#pragma unroll
                for (int e = 0; e < 16; ++e) bv[e] = (_Float16)bf[e];
            } else {
#pragma unroll
                for (int e = 0; e < 16; ++e) bv[e] = (_Float16)0.0f;
            }
            acc[nt] = __builtin_amdgcn_wmma_f32_16x16x32_f16(
                          false, av, false, bv, (short)0, acc[nt], false, false);
        }
    }
}

__global__ void wmma_gemm_tb_kernel(const float* __restrict__ A,
                                    const float* __restrict__ W,   // [N,K]
                                    float* __restrict__ C,
                                    int M, int N, int K)
{
    const int lane = threadIdx.x & 31;
    const int wave = threadIdx.x >> 5;
    const int hf   = lane >> 4;
    const int l16  = lane & 15;

    const int tilesN64 = (N + 63) >> 6;
    const int tile = blockIdx.x * 4 + wave;
    const int tM = tile / tilesN64;
    const int tN = tile - tM * tilesN64;
    if (tM * 16 >= M) return;

    const float* arow = A + (size_t)(tM * 16 + l16) * K;
    const int nStrip = tN * 64;

    v8f acc[4] = {v8f{}, v8f{}, v8f{}, v8f{}};

    if (nStrip + 64 <= N) {
        tb_accumulate<true>(arow, W, nStrip, N, K, hf, l16, acc);
#pragma unroll
        for (int nt = 0; nt < 4; ++nt) {
            const int n = nStrip + nt * 16 + l16;
#pragma unroll
            for (int r = 0; r < 8; ++r) {
                const int m = tM * 16 + r + hf * 8;
                C[(size_t)m * N + n] = acc[nt][r];
            }
        }
    } else {
        tb_accumulate<false>(arow, W, nStrip, N, K, hf, l16, acc);
#pragma unroll
        for (int nt = 0; nt < 4; ++nt) {
            const int n = nStrip + nt * 16 + l16;
            if (n < N) {
#pragma unroll
                for (int r = 0; r < 8; ++r) {
                    const int m = tM * 16 + r + hf * 8;
                    C[(size_t)m * N + n] = acc[nt][r];
                }
            }
        }
    }
}

// ---------------------------------------------------------------------------
// Embedding gather: h[row, :] = emb[x[row], :]
// ---------------------------------------------------------------------------
__global__ void embed_kernel(const int* __restrict__ x,
                             const float* __restrict__ emb,
                             float* __restrict__ h)
{
    const int row = blockIdx.x;
    const int tok = x[row];
    const float* src = emb + (size_t)tok * DIM_D;
    float* dst = h + (size_t)row * DIM_D;
    for (int i = threadIdx.x; i < DIM_D; i += blockDim.x) dst[i] = src[i];
}

// ---------------------------------------------------------------------------
// LayerNorm over last dim (D=384), one block per row.
// ---------------------------------------------------------------------------
__global__ void layernorm_kernel(const float* __restrict__ x,
                                 const float* __restrict__ g,
                                 const float* __restrict__ b,
                                 float* __restrict__ out)
{
    __shared__ float red[128];
    const int row = blockIdx.x;
    const int tid = threadIdx.x;
    const float* xr = x + (size_t)row * DIM_D;

    float s = 0.0f;
    for (int i = tid; i < DIM_D; i += 128) s += xr[i];
    red[tid] = s; __syncthreads();
    for (int o = 64; o > 0; o >>= 1) {
        if (tid < o) red[tid] += red[tid + o];
        __syncthreads();
    }
    const float mean = red[0] * (1.0f / DIM_D);
    __syncthreads();

    float v = 0.0f;
    for (int i = tid; i < DIM_D; i += 128) { float d = xr[i] - mean; v += d * d; }
    red[tid] = v; __syncthreads();
    for (int o = 64; o > 0; o >>= 1) {
        if (tid < o) red[tid] += red[tid + o];
        __syncthreads();
    }
    const float inv = rsqrtf(red[0] * (1.0f / DIM_D) + 1e-5f);

    float* orow = out + (size_t)row * DIM_D;
    for (int i = tid; i < DIM_D; i += 128)
        orow[i] = (xr[i] - mean) * inv * g[i] + b[i];
}

// ---------------------------------------------------------------------------
// Elementwise helpers
// ---------------------------------------------------------------------------
__global__ void add_inplace_kernel(float* __restrict__ dst,
                                   const float* __restrict__ src, int n)
{
    int i = blockIdx.x * blockDim.x + threadIdx.x;
    if (i < n) dst[i] += src[i];
}

__global__ void gate_u_kernel(const float* __restrict__ xn,
                              const float* __restrict__ gate,
                              const float* __restrict__ cond,
                              float* __restrict__ u, int n)
{
    int i = blockIdx.x * blockDim.x + threadIdx.x;
    if (i < n) u[i] = xn[i] * sigmoidf(gate[i]) + cond[i];
}

__global__ void og_accum_kernel(float* __restrict__ h,
                                const float* __restrict__ y,
                                const float* __restrict__ og, int n)
{
    int i = blockIdx.x * blockDim.x + threadIdx.x;
    if (i < n) h[i] += y[i] * sigmoidf(og[i]);
}

// ---------------------------------------------------------------------------
// Spectral scan: 256 independent (b,f) recurrences, sequential over T.
// ---------------------------------------------------------------------------
__global__ void spectral_scan_kernel(const float* __restrict__ spec,
                                     const float* __restrict__ decay,
                                     const float* __restrict__ freq,
                                     float* __restrict__ yc)
{
    const int id = blockIdx.x * blockDim.x + threadIdx.x;
    if (id >= SEQ_B * DIM_NF) return;
    const int bb = id / DIM_NF;
    const int f  = id - bb * DIM_NF;
    const float om = freq[f] * 0.1f;
    const float rr = cosf(om), ri = sinf(om);
    float cr = 0.0f, ci = 0.0f;
    const size_t base = (size_t)bb * SEQ_T;
    for (int t = 0; t < SEQ_T; ++t) {
        const size_t row = base + t;
        const float d  = decay[row * DIM_NF + f];
        const float ur = spec[row * (2 * DIM_NF) + f];
        const float ui = spec[row * (2 * DIM_NF) + DIM_NF + f];
        const float sr = cr * d, si = ci * d;
        const float r2 = sr * rr - si * ri;
        const float i2 = sr * ri + si * rr;
        cr = r2 + ur; ci = i2 + ui;
        yc[row * (2 * DIM_NF) + f]          = cr;
        yc[row * (2 * DIM_NF) + DIM_NF + f] = ci;
    }
}

// ---------------------------------------------------------------------------
// KL: mean(-0.5*(1+lv-mu^2-exp(lv))); loss = 0.01 * max(mean, 0.05)
// ---------------------------------------------------------------------------
__global__ void zero_kernel(float* p) { if (threadIdx.x == 0) p[0] = 0.0f; }

__global__ void kl_partial_kernel(const float* __restrict__ mu,
                                  const float* __restrict__ lv,
                                  float* __restrict__ accum, int n)
{
    __shared__ float red[256];
    float s = 0.0f;
    for (int i = blockIdx.x * blockDim.x + threadIdx.x; i < n;
         i += gridDim.x * blockDim.x) {
        const float m = mu[i], l = lv[i];
        s += -0.5f * (1.0f + l - m * m - expf(l));
    }
    red[threadIdx.x] = s; __syncthreads();
    for (int o = 128; o > 0; o >>= 1) {
        if (threadIdx.x < o) red[threadIdx.x] += red[threadIdx.x + o];
        __syncthreads();
    }
    if (threadIdx.x == 0) atomicAdd(accum, red[0]);
}

__global__ void kl_final_kernel(const float* __restrict__ accum,
                                float* __restrict__ out, float invn)
{
    if (threadIdx.x == 0)
        out[0] = 0.01f * fmaxf(accum[0] * invn, 0.05f);
}

// ---------------------------------------------------------------------------
// Host-side GEMM dispatcher (non-transposed path; M%64==0, N%64==0, K%32==0)
// ---------------------------------------------------------------------------
static void gemm_nt(hipStream_t s, int act, bool resid,
                    const float* A, const float* W, const float* bias,
                    const float* res, float* C, int M, int N, int K)
{
    dim3 g((M / 64) * (N / 64)), b(128);
    if (resid) {
        wmma_gemm_nt_kernel<0, true><<<g, b, 0, s>>>(A, W, bias, res, C, M, N, K);
    } else if (act == 1) {
        wmma_gemm_nt_kernel<1, false><<<g, b, 0, s>>>(A, W, bias, res, C, M, N, K);
    } else if (act == 2) {
        wmma_gemm_nt_kernel<2, false><<<g, b, 0, s>>>(A, W, bias, res, C, M, N, K);
    } else {
        wmma_gemm_nt_kernel<0, false><<<g, b, 0, s>>>(A, W, bias, res, C, M, N, K);
    }
}

// ---------------------------------------------------------------------------
// kernel_launch
// Input order (setup_inputs insertion order):
//  0:x  1:emb  2:mu_w 3:mu_b 4:lv_w 5:lv_b 6:up_w 7:up_b 8:lnf_g 9:lnf_b
//  then per layer (21 each): norm_g norm_b ts_w xd_w xd_b freq fs_w gate_w
//  gate_b og_w og_b ffn_g ffn_bn w1 b1 w2 b2 cp1_w cp1_b cp2_w cp2_b
// ---------------------------------------------------------------------------
extern "C" void kernel_launch(void* const* d_in, const int* in_sizes, int n_in,
                              void* d_out, int out_size, void* d_ws, size_t ws_size,
                              hipStream_t stream)
{
    (void)in_sizes; (void)n_in; (void)out_size; (void)ws_size;

    const int*   x     = (const int*)  d_in[0];
    const float* emb   = (const float*)d_in[1];
    const float* mu_w  = (const float*)d_in[2];
    const float* mu_b  = (const float*)d_in[3];
    const float* lv_w  = (const float*)d_in[4];
    const float* lv_b  = (const float*)d_in[5];
    const float* up_w  = (const float*)d_in[6];
    const float* up_b  = (const float*)d_in[7];
    const float* lnf_g = (const float*)d_in[8];
    const float* lnf_b = (const float*)d_in[9];

    struct Layer {
        const float *norm_g, *norm_b, *ts_w, *xd_w, *xd_b, *freq, *fs_w;
        const float *gate_w, *gate_b, *og_w, *og_b, *ffn_g, *ffn_bn;
        const float *w1, *b1, *w2, *b2, *cp1_w, *cp1_b, *cp2_w, *cp2_b;
    } L[N_LAYERS];
    int idx = 10;
    for (int l = 0; l < N_LAYERS; ++l) {
        L[l].norm_g = (const float*)d_in[idx++]; L[l].norm_b = (const float*)d_in[idx++];
        L[l].ts_w   = (const float*)d_in[idx++]; L[l].xd_w   = (const float*)d_in[idx++];
        L[l].xd_b   = (const float*)d_in[idx++]; L[l].freq   = (const float*)d_in[idx++];
        L[l].fs_w   = (const float*)d_in[idx++]; L[l].gate_w = (const float*)d_in[idx++];
        L[l].gate_b = (const float*)d_in[idx++]; L[l].og_w   = (const float*)d_in[idx++];
        L[l].og_b   = (const float*)d_in[idx++]; L[l].ffn_g  = (const float*)d_in[idx++];
        L[l].ffn_bn = (const float*)d_in[idx++]; L[l].w1     = (const float*)d_in[idx++];
        L[l].b1     = (const float*)d_in[idx++]; L[l].w2     = (const float*)d_in[idx++];
        L[l].b2     = (const float*)d_in[idx++]; L[l].cp1_w  = (const float*)d_in[idx++];
        L[l].cp1_b  = (const float*)d_in[idx++]; L[l].cp2_w  = (const float*)d_in[idx++];
        L[l].cp2_b  = (const float*)d_in[idx++];
    }

    float* out    = (float*)d_out;
    float* logits = out;                            // [ROWS, V]
    float* loss   = out + (size_t)ROWS * DIM_V;     // 1 float

    // Workspace carve-up (all offsets multiple of 4 floats)
    float* ws = (float*)d_ws;
    size_t off = 0;
    float* h     = ws + off; off += (size_t)ROWS * DIM_D;
    float* bott  = ws + off; off += (size_t)ROWS * DIM_D;
    float* mu    = ws + off; off += (size_t)ROWS * DIM_BD;
    float* lv    = ws + off; off += (size_t)ROWS * DIM_BD;
    float* xn    = ws + off; off += (size_t)ROWS * DIM_D;
    float* t1    = ws + off; off += (size_t)ROWS * 4 * DIM_D;  // cp1 out / ffn mid
    float* cond  = ws + off; off += (size_t)ROWS * DIM_D;
    float* gatb  = ws + off; off += (size_t)ROWS * DIM_D;      // gate / og scratch
    float* u     = ws + off; off += (size_t)ROWS * DIM_D;
    float* decay = ws + off; off += (size_t)ROWS * DIM_NF;
    float* spec  = ws + off; off += (size_t)ROWS * 2 * DIM_NF;
    float* yc    = ws + off; off += (size_t)ROWS * 2 * DIM_NF;
    float* ybuf  = ws + off; off += (size_t)ROWS * DIM_D;
    float* hn    = ws + off; off += (size_t)ROWS * DIM_D;
    float* accum = ws + off; off += 4;

    const int nD = ROWS * DIM_D;
    const dim3 eb(256);
    const dim3 egD((nD + 255) / 256);

    // 1) Embedding + bottleneck
    embed_kernel<<<ROWS, 128, 0, stream>>>(x, emb, h);
    gemm_nt(stream, 0, false, h,  mu_w, mu_b, nullptr, mu,   ROWS, DIM_BD, DIM_D);
    gemm_nt(stream, 0, false, h,  lv_w, lv_b, nullptr, lv,   ROWS, DIM_BD, DIM_D);
    gemm_nt(stream, 0, false, mu, up_w, up_b, nullptr, bott, ROWS, DIM_D,  DIM_BD);
    add_inplace_kernel<<<egD, eb, 0, stream>>>(h, bott, nD);

    // 2) Layers
    for (int l = 0; l < N_LAYERS; ++l) {
        // cond = gelu(bott @ cp1 + b) @ cp2 + b
        gemm_nt(stream, 1, false, bott, L[l].cp1_w, L[l].cp1_b, nullptr, t1,   ROWS, DIM_D, DIM_D);
        gemm_nt(stream, 0, false, t1,   L[l].cp2_w, L[l].cp2_b, nullptr, cond, ROWS, DIM_D, DIM_D);
        // xn = LN(h)
        layernorm_kernel<<<ROWS, 128, 0, stream>>>(h, L[l].norm_g, L[l].norm_b, xn);
        // u = xn * sigmoid(xn @ gate_w + gate_b) + cond
        gemm_nt(stream, 0, false, xn, L[l].gate_w, L[l].gate_b, nullptr, gatb, ROWS, DIM_D, DIM_D);
        gate_u_kernel<<<egD, eb, 0, stream>>>(xn, gatb, cond, u, nD);
        // decay = sigmoid(u @ xd_w + xd_b); spectral = u @ ts_w
        gemm_nt(stream, 2, false, u, L[l].xd_w, L[l].xd_b, nullptr, decay, ROWS, DIM_NF,     DIM_D);
        gemm_nt(stream, 0, false, u, L[l].ts_w, nullptr,  nullptr, spec,  ROWS, 2 * DIM_NF, DIM_D);
        // sequential complex scan
        spectral_scan_kernel<<<1, 256, 0, stream>>>(spec, decay, L[l].freq, yc);
        // y = yc @ fs_w ; og = xn @ og_w + og_b ; h += y * sigmoid(og)
        gemm_nt(stream, 0, false, yc, L[l].fs_w, nullptr,   nullptr, ybuf, ROWS, DIM_D, 2 * DIM_NF);
        gemm_nt(stream, 0, false, xn, L[l].og_w, L[l].og_b, nullptr, gatb, ROWS, DIM_D, DIM_D);
        og_accum_kernel<<<egD, eb, 0, stream>>>(h, ybuf, gatb, nD);
        // FFN: h += gelu(LN(h) @ w1 + b1) @ w2 + b2
        layernorm_kernel<<<ROWS, 128, 0, stream>>>(h, L[l].ffn_g, L[l].ffn_bn, hn);
        gemm_nt(stream, 1, false, hn, L[l].w1, L[l].b1, nullptr, t1, ROWS, 4 * DIM_D, DIM_D);
        gemm_nt(stream, 0, true,  t1, L[l].w2, L[l].b2, h,       h,  ROWS, DIM_D, 4 * DIM_D);
    }

    // 3) Final LN + logits = hn @ emb^T   (HBM-bound: single pass over output)
    layernorm_kernel<<<ROWS, 128, 0, stream>>>(h, lnf_g, lnf_b, hn);
    {
        const int tilesN64 = (DIM_V + 63) / 64;          // 786
        const int waveTiles = (ROWS / 16) * tilesN64;    // 256 * 786
        dim3 g((waveTiles + 3) / 4), b(128);
        wmma_gemm_tb_kernel<<<g, b, 0, stream>>>(hn, emb, logits, ROWS, DIM_V, DIM_D);
    }

    // 4) KL regularization loss
    zero_kernel<<<1, 32, 0, stream>>>(accum);
    kl_partial_kernel<<<256, 256, 0, stream>>>(mu, lv, accum, ROWS * DIM_BD);
    kl_final_kernel<<<1, 32, 0, stream>>>(accum, loss, 1.0f / (float)(ROWS * DIM_BD));
}
